// DaeMon_84
// MI455X (gfx1250) — matrix operand order repaired and checked by reading, not compiled, not run on hardware
//
#include <hip/hip_runtime.h>
#include <hip/hip_bf16.h>

#define NNODES   5000
#define NREL2    400
#define DIN      64
#define BATCH    16
#define NEG1     32
#define NEDGES   30000
#define KTOT     832      // 13 * 64
#define KCCOUNT  26       // 832 / 32
#define EPSV     1e-6f

typedef _Float16 v8h  __attribute__((ext_vector_type(8)));
typedef _Float16 v16h __attribute__((ext_vector_type(16)));
typedef float    v8f  __attribute__((ext_vector_type(8)));

// ---------- float atomic max/min via order-preserving int trick ----------
__device__ __forceinline__ void atomicMaxF(float* a, float v) {
    if (v >= 0.0f) atomicMax((int*)a, __float_as_int(v));
    else           atomicMin((unsigned int*)a, __float_as_uint(v));
}
__device__ __forceinline__ void atomicMinF(float* a, float v) {
    if (v >= 0.0f) atomicMin((int*)a, __float_as_int(v));
    else           atomicMax((unsigned int*)a, __float_as_uint(v));
}

// ---------- query / index preparation ----------
__global__ void prep_kernel(const int* __restrict__ qt, const float* __restrict__ qw,
                            int* __restrict__ h_i0, int* __restrict__ t_i,
                            float* __restrict__ query) {
    int b = threadIdx.x;
    if (b < BATCH) {
        int h0 = qt[(b * NEG1 + 0) * 3 + 0];
        bool neg = true;
        for (int j = 0; j < NEG1; ++j) neg &= (qt[(b * NEG1 + j) * 3 + 0] == h0);
        int hh = qt[(b * NEG1) * 3 + 0];
        int tt = qt[(b * NEG1) * 3 + 2];
        int rr = qt[(b * NEG1) * 3 + 1];
        h_i0[b] = neg ? hh : tt;
        int r = neg ? rr : rr + NREL2 / 2;
        for (int j = 0; j < NEG1; ++j) {
            int hj = qt[(b * NEG1 + j) * 3 + 0];
            int tj = qt[(b * NEG1 + j) * 3 + 2];
            t_i[b * NEG1 + j] = neg ? tj : hj;
        }
        for (int d = 0; d < DIN; ++d) query[b * DIN + d] = qw[r * DIN + d];
    }
}

// ---------- pack layer weights into WMMA B-operand layout (f16) ----------
// Wpack[li][kc][nt][lane][i]: k = kc*32 + (lane/16)*16 + i, col = nt*16 + lane%16
__global__ void pack_w_kernel(const float* __restrict__ Ws, _Float16* __restrict__ wpack) {
    int idx = blockIdx.x * 256 + threadIdx.x;
    const int TOT = 2 * KCCOUNT * 4 * 32 * 16;
    if (idx >= TOT) return;
    int i  = idx % 16;
    int t1 = idx / 16;
    int L  = t1 % 32;
    int t2 = t1 / 32;
    int nt = t2 % 4;
    int t3 = t2 / 4;
    int kc = t3 % KCCOUNT;
    int li = t3 / KCCOUNT;
    int k = kc * 32 + (L / 16) * 16 + i;
    int c = nt * 16 + (L % 16);
    wpack[idx] = (_Float16)Ws[((size_t)li * KTOT + k) * DIN + c];
}

// ---------- boundary / initial state: (N,B,D) node-major ----------
__global__ void init_initial_kernel(const float* __restrict__ query, const int* __restrict__ h_i0,
                                    const float* __restrict__ prev, int snap,
                                    float* __restrict__ init) {
    size_t g = (size_t)blockIdx.x * 256 + threadIdx.x;
    if (g >= (size_t)NNODES * BATCH * DIN) return;
    int n = (int)(g / (BATCH * DIN));
    int rem = (int)(g % (BATCH * DIN));
    int m = rem / DIN, d = rem % DIN;
    float v = (n == h_i0[m]) ? query[m * DIN + d] : 0.0f;
    if (snap) v = 0.5f * (v + prev[g]);
    init[g] = v;
}

// ---------- accumulator init with the boundary self-message ----------
__global__ void init_acc_kernel(const float* __restrict__ bnd,
                                float* __restrict__ sm, float* __restrict__ sq,
                                float* __restrict__ mx, float* __restrict__ mn) {
    size_t g = (size_t)blockIdx.x * 256 + threadIdx.x;
    if (g >= (size_t)NNODES * BATCH * DIN) return;
    float v = bnd[g];
    sm[g] = v; sq[g] = v * v; mx[g] = v; mn[g] = v;
}

// ---------- degree ----------
__global__ void deg_init_kernel(int* __restrict__ deg) {
    int n = blockIdx.x * 256 + threadIdx.x;
    if (n < NNODES) deg[n] = 1;   // boundary self-edge
}
__global__ void deg_count_kernel(const int* __restrict__ dst, int* __restrict__ deg) {
    int e = blockIdx.x * 256 + threadIdx.x;
    if (e < NEDGES) atomicAdd(&deg[dst[e]], 1);
}

// ---------- scale = log(deg)/mean(log(deg)); one block reduction ----------
__global__ void scale_kernel(const int* __restrict__ deg, float* __restrict__ scale,
                             float* __restrict__ invscale, float* __restrict__ invdeg) {
    __shared__ float red[256];
    float local = 0.0f;
    for (int n = threadIdx.x; n < NNODES; n += 256) local += __logf((float)deg[n]);
    red[threadIdx.x] = local;
    __syncthreads();
    for (int s = 128; s > 0; s >>= 1) {
        if (threadIdx.x < s) red[threadIdx.x] += red[threadIdx.x + s];
        __syncthreads();
    }
    float meanlog = red[0] / (float)NNODES;
    for (int n = threadIdx.x; n < NNODES; n += 256) {
        float dg = (float)deg[n];
        float sc = __logf(dg) / meanlog;
        scale[n] = sc;
        float cl = sc < 0.01f ? 0.01f : sc;
        invscale[n] = 1.0f / cl;
        invdeg[n] = 1.0f / dg;
    }
}

// ---------- edge aggregation: L2-resident atomics (82 MB fits in 192 MB L2) ----------
__global__ __launch_bounds__(256) void edge_agg_kernel(
        const int* __restrict__ src, const int* __restrict__ dst, const int* __restrict__ et,
        const float* __restrict__ x, const float* __restrict__ rel,
        float* __restrict__ sm, float* __restrict__ sq,
        float* __restrict__ mx, float* __restrict__ mn) {
    int e = blockIdx.x;
    int s = src[e], d0 = dst[e], ty = et[e];
    const float* xrow = x + (size_t)s * BATCH * DIN;
    float* sb = sm + (size_t)d0 * BATCH * DIN;
    float* qb = sq + (size_t)d0 * BATCH * DIN;
    float* xb = mx + (size_t)d0 * BATCH * DIN;
    float* nb = mn + (size_t)d0 * BATCH * DIN;
    const float* rrow = rel + (size_t)ty * DIN;
#pragma unroll
    for (int k = 0; k < (BATCH * DIN) / 256; ++k) {
        int p = threadIdx.x + k * 256;          // p = m*64 + d (contiguous)
        int d = p & (DIN - 1);
        float v = xrow[p] * rrow[d];
        atomicAdd(&sb[p], v);
        atomicAdd(&qb[p], v * v);
        atomicMaxF(&xb[p], v);
        atomicMinF(&nb[p], v);
    }
}

// ---------- fused PNA feature build + WMMA GEMM (per node: M=16 batches) ----------
__global__ __launch_bounds__(64) void pna_gemm_kernel(
        const float* __restrict__ x,
        const float* __restrict__ sm, const float* __restrict__ sq,
        const float* __restrict__ mx, const float* __restrict__ mn,
        const float* __restrict__ invdeg, const float* __restrict__ scale,
        const float* __restrict__ invscale,
        const _Float16* __restrict__ wpack,     // (26,4,32,16) f16, layer base
        const float* __restrict__ bias,         // (64)
        float* __restrict__ xout) {
    __shared__ _Float16 Atile[2][16 * KTOT];    // 53,248 B
    int wave = threadIdx.x >> 5;
    int lane = threadIdx.x & 31;
    int n = blockIdx.x * 2 + wave;
    if (n >= NNODES) return;                    // never taken (2500*2 == 5000)

    float idg = invdeg[n], sc = scale[n], isc = invscale[n];
    _Float16* A = Atile[wave];

    // build 16x832 A tile: [x(64) | (d*4+f)*3+s features(768)]
    for (int p = lane; p < BATCH * DIN; p += 32) {
        int m = p >> 6, d = p & (DIN - 1);
        size_t g = ((size_t)n * BATCH + m) * DIN + d;
        float mean = sm[g] * idg;
        float var  = sq[g] * idg - mean * mean;
        var = var < EPSV ? EPSV : var;
        float stdv = __fsqrt_rn(var);
        float fx = mx[g], fn = mn[g];
        _Float16* row = A + m * KTOT;
        row[d] = (_Float16)x[g];
        int base = DIN + d * 12;
        row[base + 0]  = (_Float16)mean;
        row[base + 1]  = (_Float16)(mean * sc);
        row[base + 2]  = (_Float16)(mean * isc);
        row[base + 3]  = (_Float16)fx;
        row[base + 4]  = (_Float16)(fx * sc);
        row[base + 5]  = (_Float16)(fx * isc);
        row[base + 6]  = (_Float16)fn;
        row[base + 7]  = (_Float16)(fn * sc);
        row[base + 8]  = (_Float16)(fn * isc);
        row[base + 9]  = (_Float16)stdv;
        row[base + 10] = (_Float16)(stdv * sc);
        row[base + 11] = (_Float16)(stdv * isc);
    }
    __syncthreads();

    // A operand: lane L -> row L%16; per kc lane holds K = {koff..koff+15}
    const _Float16* Arow = A + (lane & 15) * KTOT;
    int koff = (lane >> 4) * 16;

    for (int nt = 0; nt < 4; ++nt) {
        v8f acc = {};
#pragma unroll
        for (int kc = 0; kc < KCCOUNT; ++kc) {
            v8h lo = *(const v8h*)(Arow + kc * 32 + koff);
            v8h hi = *(const v8h*)(Arow + kc * 32 + koff + 8);
            v16h av = __builtin_shufflevector(lo, hi, 0,1,2,3,4,5,6,7,8,9,10,11,12,13,14,15);
            v16h bv = *(const v16h*)(wpack + ((size_t)(kc * 4 + nt) * 32 + lane) * 16);
            acc = __builtin_amdgcn_wmma_f32_16x16x32_f16(
                      false, av, false, bv, (short)0, acc, false, false);
        }
        int c = nt * 16 + (lane & 15);
        float bv = bias[c];
#pragma unroll
        for (int r = 0; r < 8; ++r) {
            int m = (lane >> 4) * 8 + r;       // D layout: lanes 0-15 rows 0-7, lanes 16-31 rows 8-15
            float v = acc[r] + bv;
            v = v > 0.0f ? v : 0.0f;
            xout[((size_t)n * BATCH + m) * DIN + c] = v;
        }
    }
}

// ---------- final scorer: feat(128) -> relu(W1) -> W2 ----------
__global__ __launch_bounds__(128) void score_kernel(
        const float* __restrict__ out, const float* __restrict__ query,
        const int* __restrict__ t_i,
        const float* __restrict__ w1, const float* __restrict__ b1,
        const float* __restrict__ w2, const float* __restrict__ b2,
        float* __restrict__ score) {
    __shared__ float feat[128];
    __shared__ float hid[128];
    int bj = blockIdx.x;
    int b = bj / NEG1;
    int t = t_i[bj];
    int tid = threadIdx.x;
    if (tid < 64) feat[tid] = out[((size_t)t * BATCH + b) * DIN + tid];
    else          feat[tid] = query[b * DIN + (tid - 64)];
    __syncthreads();
    float a = b1[tid];
    for (int k = 0; k < 128; ++k) a += feat[k] * w1[k * 128 + tid];
    a = a > 0.0f ? a : 0.0f;
    hid[tid] = a * w2[tid];
    __syncthreads();
    for (int s = 64; s > 0; s >>= 1) {
        if (tid < s) hid[tid] += hid[tid + s];
        __syncthreads();
    }
    if (tid == 0) score[bj] = hid[0] + b2[0];
}

extern "C" void kernel_launch(void* const* d_in, const int* in_sizes, int n_in,
                              void* d_out, int out_size, void* d_ws, size_t ws_size,
                              hipStream_t stream) {
    (void)in_sizes; (void)n_in; (void)out_size; (void)ws_size;
    const int*   qt   = (const int*)d_in[0];     // (16,32,3)
    const int*   ei   = (const int*)d_in[1];     // (2,2,30000)
    const int*   etp  = (const int*)d_in[2];     // (2,30000)
    const float* qw   = (const float*)d_in[3];   // (400,64)
    const float* rel  = (const float*)d_in[4];   // (2,400,64)
    const float* Ws   = (const float*)d_in[5];   // (2,832,64)
    const float* bs   = (const float*)d_in[6];   // (2,64)
    const float* w1   = (const float*)d_in[7];   // (128,128)
    const float* b1   = (const float*)d_in[8];   // (128)
    const float* w2   = (const float*)d_in[9];   // (128,1)
    const float* b2   = (const float*)d_in[10];  // (1)

    const size_t BUFB = (size_t)NNODES * BATCH * DIN * sizeof(float); // 20.48 MB
    char* wsp = (char*)d_ws;
    auto take = [&](size_t bytes) { char* p = wsp; wsp += (bytes + 255) & ~(size_t)255; return p; };

    float*     query    = (float*)take(BATCH * DIN * sizeof(float));
    int*       h_i0     = (int*)take(BATCH * sizeof(int));
    int*       t_i      = (int*)take(BATCH * NEG1 * sizeof(int));
    int*       deg      = (int*)take(NNODES * sizeof(int));
    float*     scale    = (float*)take(NNODES * sizeof(float));
    float*     invscale = (float*)take(NNODES * sizeof(float));
    float*     invdeg   = (float*)take(NNODES * sizeof(float));
    _Float16*  wpack    = (_Float16*)take(2 * KCCOUNT * 4 * 32 * 16 * sizeof(_Float16));
    float*     bufInit  = (float*)take(BUFB);
    float*     bufA     = (float*)take(BUFB);
    float*     bufB     = (float*)take(BUFB);
    float*     accSum   = (float*)take(BUFB);
    float*     accSq    = (float*)take(BUFB);
    float*     accMx    = (float*)take(BUFB);
    float*     accMn    = (float*)take(BUFB);

    const int NBD = NNODES * BATCH * DIN;
    const int nbdBlocks = (NBD + 255) / 256;

    prep_kernel<<<1, 64, 0, stream>>>(qt, qw, h_i0, t_i, query);
    pack_w_kernel<<<(2 * KCCOUNT * 4 * 32 * 16 + 255) / 256, 256, 0, stream>>>(Ws, wpack);

    for (int snap = 0; snap < 2; ++snap) {
        const int* src = ei + (size_t)(snap * 2 + 0) * NEDGES;
        const int* dst = ei + (size_t)(snap * 2 + 1) * NEDGES;
        const int* et  = etp + (size_t)snap * NEDGES;

        init_initial_kernel<<<nbdBlocks, 256, 0, stream>>>(query, h_i0, bufB, snap, bufInit);
        deg_init_kernel<<<(NNODES + 255) / 256, 256, 0, stream>>>(deg);
        deg_count_kernel<<<(NEDGES + 255) / 256, 256, 0, stream>>>(dst, deg);
        scale_kernel<<<1, 256, 0, stream>>>(deg, scale, invscale, invdeg);

        const float* xcur = bufInit;
        float* xnext = bufA;
        for (int li = 0; li < 2; ++li) {
            init_acc_kernel<<<nbdBlocks, 256, 0, stream>>>(bufInit, accSum, accSq, accMx, accMn);
            edge_agg_kernel<<<NEDGES, 256, 0, stream>>>(src, dst, et, xcur,
                    rel + (size_t)li * NREL2 * DIN, accSum, accSq, accMx, accMn);
            pna_gemm_kernel<<<NNODES / 2, 64, 0, stream>>>(xcur, accSum, accSq, accMx, accMn,
                    invdeg, scale, invscale,
                    wpack + (size_t)li * KCCOUNT * 4 * 32 * 16,
                    bs + (size_t)li * DIN, xnext);
            xcur = xnext;
            xnext = (li == 0) ? bufB : bufA;
        }
        // after layer 1, output lives in bufB (read by next snapshot's init)
    }

    score_kernel<<<BATCH * NEG1, 128, 0, stream>>>(bufB, query, t_i, w1, b1, w2, b2,
                                                   (float*)d_out);
}
